// AttentionBlock_33913061769296
// MI455X (gfx1250) — compile-verified, matrix operand order checked by reading, not run on hardware
//
#include <hip/hip_runtime.h>
#include <hip/hip_bf16.h>
#include <math.h>

// ---------- CDNA5 WMMA plumbing ----------
typedef __attribute__((ext_vector_type(16))) _Float16 v16h;
typedef __attribute__((ext_vector_type(8)))  _Float16 v8h;
typedef __attribute__((ext_vector_type(8)))  float    v8f;

// Optional CDNA5 async global->LDS copy path (ASYNCcnt).
#if defined(__has_builtin)
#  if __has_builtin(__builtin_amdgcn_global_load_async_to_lds_b128) && \
      __has_builtin(__builtin_amdgcn_s_wait_asynccnt)
#    define HAVE_ASYNC_LDS 1
#  endif
#endif
#ifndef HAVE_ASYNC_LDS
#  define HAVE_ASYNC_LDS 0
#endif

#if HAVE_ASYNC_LDS
// Builtin signature (from hipcc diagnostic): param 1 is
//   int __attribute__((vector_size(16))) addrspace(1)*  (non-const)
typedef int b128_t __attribute__((vector_size(16)));
typedef b128_t __attribute__((address_space(1)))* gas_b128p;
typedef b128_t __attribute__((address_space(3)))* las_b128p;
#endif

// Load a 16x32 (A, row=M) or 32x16-column-major (B, row=N) f16 fragment from a
// row-major tile with leading dimension `ld` (in halves, multiple of 8).
// Per ISA 7.12.2: lanes 0-15 hold K {0..7,16..23}, lanes 16-31 hold K {8..15,24..31}.
__device__ __forceinline__ v16h load_frag(const _Float16* __restrict__ base, int ld) {
  const int lane = threadIdx.x & 31;
  const int row  = lane & 15;
  const int hi   = lane >> 4;
  const _Float16* p = base + (size_t)row * ld + hi * 8;
  v8h c0 = *(const v8h*)(p);        // K = hi*8 .. hi*8+7
  v8h c1 = *(const v8h*)(p + 16);   // K = 16 + hi*8 ..
  v16h r;
#pragma unroll
  for (int i = 0; i < 8; ++i) { r[i] = c0[i]; r[i + 8] = c1[i]; }
  return r;
}

__device__ __forceinline__ v8f wmma16(v16h a, v16h b, v8f c) {
  // D = A(16x32) * B(32x16) + C, f32 accumulate
  return __builtin_amdgcn_wmma_f32_16x16x32_f16(false, a, false, b, (short)0, c,
                                                false, false);
}

// ---------- constants ----------
#define BATCH 32
#define CH    256
#define HW    1024
#define GRP   8
#define CPG   32          // channels per group
#define SCALE 0.0625f     // 1/sqrt(256)

// ---------- kernel 0: weights f32 -> f16 ----------
__global__ __launch_bounds__(256) void cvt_weights_kernel(
    const float* __restrict__ wq, const float* __restrict__ wp,
    _Float16* __restrict__ wq16, _Float16* __restrict__ wp16) {
  int i = blockIdx.x * 256 + threadIdx.x;
  if (i < 3 * CH * CH) wq16[i] = (_Float16)wq[i];
  if (i < CH * CH)     wp16[i] = (_Float16)wp[i];
}

// ---------- kernel 1: GroupNorm statistics ----------
__global__ __launch_bounds__(256) void gn_stats_kernel(
    const float* __restrict__ x, float* __restrict__ mean, float* __restrict__ rstd) {
  const int bg = blockIdx.x;                  // b*8+g
  const float* xp = x + (size_t)bg * CPG * HW;
  float s = 0.f, s2 = 0.f;
  for (int i = threadIdx.x; i < CPG * HW; i += 256) {
    float v = xp[i]; s += v; s2 += v * v;
  }
  __shared__ float sm[256], sq[256];
  sm[threadIdx.x] = s; sq[threadIdx.x] = s2;
  __syncthreads();
  for (int off = 128; off > 0; off >>= 1) {
    if (threadIdx.x < off) {
      sm[threadIdx.x] += sm[threadIdx.x + off];
      sq[threadIdx.x] += sq[threadIdx.x + off];
    }
    __syncthreads();
  }
  if (threadIdx.x == 0) {
    const float inv_n = 1.0f / (CPG * HW);
    float m = sm[0] * inv_n;
    float var = sq[0] * inv_n - m * m;
    mean[bg] = m;
    rstd[bg] = rsqrtf(var + 1e-5f);
  }
}

// ---------- kernel 2: GN-fused QKV GEMM ----------
// Block tile: 64 pixels x 64 outputs. 8 waves as 2x4; each wave: 32x16 (2 WMMA/K-step).
// Each K-step (32 channels) is exactly one GN group. A-tile double-buffered in LDS:
// one barrier per K-step; each thread stages 8 adjacent channels -> 1 ds_store_b128.
__global__ __launch_bounds__(256) void qkv_gemm_kernel(
    const float* __restrict__ x, const float* __restrict__ gamma,
    const float* __restrict__ beta, const float* __restrict__ mean,
    const float* __restrict__ rstd, const _Float16* __restrict__ wq16,
    const float* __restrict__ b_qkv,
    _Float16* __restrict__ q16, _Float16* __restrict__ k16,
    _Float16* __restrict__ vt16) {
  const int o0 = blockIdx.x * 64;   // 0..704
  const int p0 = blockIdx.y * 64;   // 0..960
  const int b  = blockIdx.z;
  const int tid = threadIdx.x, wave = tid >> 5, lane = tid & 31;
  const int wm = wave >> 2, wn = wave & 3;
  const int n = lane & 15, hi = lane >> 4;

  __shared__ _Float16 At[2][64 * 40];  // [pixel][channel], ld=40 halves

  // staging: thread t -> pixel p = t&63, channels cg..cg+7 (one b128 LDS store)
  const int sp = tid & 63;
  const int scg = (tid >> 6) * 8;      // 0,8,16,24
  auto stage = [&](int kk, int buf) {
    const int c0 = kk * CPG;
    const float mu = mean[b * GRP + kk];
    const float rs = rstd[b * GRP + kk];
    v8h pk;
#pragma unroll
    for (int i = 0; i < 8; ++i) {
      int c = scg + i;
      float v = x[((size_t)(b * CH + c0 + c)) * HW + p0 + sp];
      pk[i] = (_Float16)((v - mu) * rs * gamma[c0 + c] + beta[c0 + c]);
    }
    *(v8h*)(&At[buf][sp * 40 + scg]) = pk;
  };

  v8f acc0 = {}, acc1 = {};
  stage(0, 0);
  for (int kk = 0; kk < 8; ++kk) {
    __syncthreads();
    if (kk < 7) stage(kk + 1, (kk + 1) & 1);
    const _Float16* cur = At[kk & 1];
    v16h bf = load_frag(wq16 + (size_t)(o0 + wn * 16) * CH + kk * CPG, CH);
    v16h a0 = load_frag(cur + (wm * 32) * 40, 40);
    v16h a1 = load_frag(cur + (wm * 32 + 16) * 40, 40);
    acc0 = wmma16(a0, bf, acc0);
    acc1 = wmma16(a1, bf, acc1);
  }

  const int oo = o0 + wn * 16 + n;
  const float bias = b_qkv[oo];
  if (oo < 2 * CH) {                  // Q,K: (pixel, channel) layout
    _Float16* dst = (oo < CH) ? q16 : k16;
    const int cc = (oo < CH) ? oo : oo - CH;
#pragma unroll
    for (int r = 0; r < 8; ++r) {
      int m0 = p0 + wm * 32 + hi * 8 + r;
      dst[(((size_t)b << 10) + m0) * CH + cc]      = (_Float16)(acc0[r] + bias);
      dst[(((size_t)b << 10) + m0 + 16) * CH + cc] = (_Float16)(acc1[r] + bias);
    }
  } else {                            // V transposed: (channel, pixel) -> contiguous
    const int vrow = oo - 2 * CH;
    size_t base = ((size_t)b * CH + vrow) * HW + p0 + wm * 32 + hi * 8;
    v8h pk0, pk1;
#pragma unroll
    for (int r = 0; r < 8; ++r) {
      pk0[r] = (_Float16)(acc0[r] + bias);
      pk1[r] = (_Float16)(acc1[r] + bias);
    }
    *(v8h*)(vt16 + base)      = pk0;
    *(v8h*)(vt16 + base + 16) = pk1;
  }
}

// ---------- kernel 3: flash attention (16 q-rows per block, online softmax) ----------
__global__ __launch_bounds__(256) void attn_kernel(
    const _Float16* __restrict__ q16, const _Float16* __restrict__ k16,
    const _Float16* __restrict__ vt16, _Float16* __restrict__ h2) {
  const int q0 = blockIdx.x * 16;
  const int b  = blockIdx.y;
  const int tid = threadIdx.x, wave = tid >> 5, lane = tid & 31;
  const int n = lane & 15, hi = lane >> 4;
  const int srow = tid >> 4, sseg = tid & 15;   // softmax mapping: 16 thr/row

  __shared__ _Float16 Qt[16 * 264];   // Q tile (pixel, channel), ld=264
  __shared__ float    St[16 * 132];   // scores tile 16 x 128, ld=132
  __shared__ _Float16 Pt[16 * 136];   // exp tile f16, ld=136
  __shared__ float    red[16 * 16];   // per-row partial reductions
  __shared__ float    mrun[16], lrun[16], fac[16], mnew_s[16];

  // ---- load Q tile (pure global->LDS copy: use CDNA5 async path if present) ----
#if HAVE_ASYNC_LDS
  {
#pragma unroll
    for (int it = 0; it < 2; ++it) {
      int idx = tid + it * 256;               // 0..511 chunk id (16B chunks)
      int row = idx >> 5, col = (idx & 31) * 8;
      const _Float16* src = q16 + (((size_t)b << 10) + q0 + row) * CH + col;
      _Float16* dst = Qt + row * 264 + col;
      __builtin_amdgcn_global_load_async_to_lds_b128(
          (gas_b128p)src, (las_b128p)dst, 0, 0);
    }
    __builtin_amdgcn_s_wait_asynccnt(0);
  }
#else
  {
    int p = tid >> 4, c = (tid & 15) * 16;
    const _Float16* src = q16 + (((size_t)b << 10) + q0 + p) * CH + c;
    *(v8h*)(Qt + p * 264 + c)     = *(const v8h*)(src);
    *(v8h*)(Qt + p * 264 + c + 8) = *(const v8h*)(src + 8);
  }
#endif
  if (tid < 16) { mrun[tid] = -1e30f; lrun[tid] = 0.f; }
  __syncthreads();

  v8f oa = {}, ob = {};               // O accum: this wave's channels [wave*32, +32)
  const int cbase = wave * 32;

  for (int jt = 0; jt < 8; ++jt) {
    const int j0 = jt * 128;
    // --- S = Q * K^T * scale ; wave computes columns j0 + wave*16 .. +15 ---
    v8f s = {};
    const _Float16* kb = k16 + (((size_t)b << 10) + j0 + wave * 16) * CH;
#pragma unroll
    for (int kk = 0; kk < 8; ++kk) {
      v16h a  = load_frag(Qt + kk * 32, 264);
      v16h bf = load_frag(kb + kk * 32, CH);
      s = wmma16(a, bf, s);
    }
#pragma unroll
    for (int r = 0; r < 8; ++r)
      St[(r + 8 * hi) * 132 + wave * 16 + n] = s[r] * SCALE;
    __syncthreads();

    // --- online softmax, 16 threads per row ---
    {
      float lm = -1e30f;
#pragma unroll
      for (int c = 0; c < 8; ++c)
        lm = fmaxf(lm, St[srow * 132 + sseg * 8 + c]);
      red[srow * 16 + sseg] = lm;
    }
    __syncthreads();
    if (tid < 16) {
      float rmax = -1e30f;
#pragma unroll
      for (int t = 0; t < 16; ++t) rmax = fmaxf(rmax, red[tid * 16 + t]);
      float mold = mrun[tid];
      float mnew = fmaxf(mold, rmax);
      fac[tid] = __expf(mold - mnew);
      mnew_s[tid] = mnew;
      mrun[tid] = mnew;
    }
    __syncthreads();
    {
      float mnew = mnew_s[srow];
      float sum = 0.f;
#pragma unroll
      for (int c = 0; c < 8; ++c) {
        int j = sseg * 8 + c;
        float e = __expf(St[srow * 132 + j] - mnew);
        Pt[srow * 136 + j] = (_Float16)e;
        sum += e;
      }
      red[srow * 16 + sseg] = sum;
    }
    __syncthreads();
    if (tid < 16) {
      float sum = 0.f;
#pragma unroll
      for (int t = 0; t < 16; ++t) sum += red[tid * 16 + t];
      lrun[tid] = lrun[tid] * fac[tid] + sum;
    }
    __syncthreads();

    // --- rescale O, accumulate P * V ---
#pragma unroll
    for (int r = 0; r < 8; ++r) {
      float f = fac[r + 8 * hi];
      oa[r] *= f; ob[r] *= f;
    }
    const _Float16* vb = vt16 + ((size_t)b * CH + cbase) * HW + j0;
#pragma unroll
    for (int kk = 0; kk < 4; ++kk) {
      v16h a  = load_frag(Pt + kk * 32, 136);
      v16h b0 = load_frag(vb + kk * 32, HW);
      v16h b1 = load_frag(vb + (size_t)16 * HW + kk * 32, HW);
      oa = wmma16(a, b0, oa);
      ob = wmma16(a, b1, ob);
    }
    __syncthreads();
  }

#pragma unroll
  for (int r = 0; r < 8; ++r) {
    int m = r + 8 * hi;
    float inv = 1.0f / lrun[m];
    size_t rowbase = (((size_t)b << 10) + q0 + m) * CH + cbase;
    h2[rowbase + n]      = (_Float16)(oa[r] * inv);
    h2[rowbase + 16 + n] = (_Float16)(ob[r] * inv);
  }
}

// ---------- kernel 4: proj GEMM + bias + residual (frags direct from global) ----------
__global__ __launch_bounds__(256) void proj_kernel(
    const _Float16* __restrict__ h2, const _Float16* __restrict__ wp16,
    const float* __restrict__ b_proj, const float* __restrict__ x,
    float* __restrict__ out) {
  const int o0 = blockIdx.x * 64;
  const int p0 = blockIdx.y * 64;
  const int b  = blockIdx.z;
  const int tid = threadIdx.x, wave = tid >> 5, lane = tid & 31;
  const int wm = wave >> 2, wn = wave & 3;
  const int n = lane & 15, hi = lane >> 4;

  v8f acc0 = {}, acc1 = {};
  const _Float16* abase = h2 + (((size_t)b << 10) + p0 + wm * 32) * CH;
  const _Float16* bbase = wp16 + (size_t)(o0 + wn * 16) * CH;
#pragma unroll
  for (int kk = 0; kk < 8; ++kk) {
    if (kk < 7) __builtin_prefetch(abase + (kk + 1) * 32, 0, 1);
    v16h bf = load_frag(bbase + kk * 32, CH);
    v16h a0 = load_frag(abase + kk * 32, CH);
    v16h a1 = load_frag(abase + (size_t)16 * CH + kk * 32, CH);
    acc0 = wmma16(a0, bf, acc0);
    acc1 = wmma16(a1, bf, acc1);
  }
  const int oo = o0 + wn * 16 + n;
  const float bias = b_proj[oo];
  // lane's outputs are pixel-contiguous: vectorize residual add + store as float4
  size_t base = ((size_t)(b * CH + oo)) * HW + p0 + wm * 32 + hi * 8;
  const float4* xp4 = (const float4*)(x + base);
  float4* op4 = (float4*)(out + base);
  float4 x0 = xp4[0], x1 = xp4[1], x2 = xp4[4], x3 = xp4[5];
  float4 r0 = make_float4(x0.x + acc0[0] + bias, x0.y + acc0[1] + bias,
                          x0.z + acc0[2] + bias, x0.w + acc0[3] + bias);
  float4 r1 = make_float4(x1.x + acc0[4] + bias, x1.y + acc0[5] + bias,
                          x1.z + acc0[6] + bias, x1.w + acc0[7] + bias);
  float4 r2 = make_float4(x2.x + acc1[0] + bias, x2.y + acc1[1] + bias,
                          x2.z + acc1[2] + bias, x2.w + acc1[3] + bias);
  float4 r3 = make_float4(x3.x + acc1[4] + bias, x3.y + acc1[5] + bias,
                          x3.z + acc1[6] + bias, x3.w + acc1[7] + bias);
  op4[0] = r0; op4[1] = r1; op4[4] = r2; op4[5] = r3;
}

// ---------- launcher ----------
extern "C" void kernel_launch(void* const* d_in, const int* in_sizes, int n_in,
                              void* d_out, int out_size, void* d_ws, size_t ws_size,
                              hipStream_t stream) {
  const float* x      = (const float*)d_in[0];
  const float* gamma  = (const float*)d_in[1];
  const float* beta   = (const float*)d_in[2];
  const float* w_qkv  = (const float*)d_in[3];
  const float* b_qkv  = (const float*)d_in[4];
  const float* w_proj = (const float*)d_in[5];
  const float* b_proj = (const float*)d_in[6];
  float* out = (float*)d_out;

  char* ws = (char*)d_ws;
  _Float16* wq16 = (_Float16*)ws; ws += (size_t)3 * CH * CH * 2;   // 384 KB
  _Float16* wp16 = (_Float16*)ws; ws += (size_t)CH * CH * 2;       // 128 KB
  float*    mean = (float*)ws;    ws += BATCH * GRP * 4;
  float*    rstd = (float*)ws;    ws += BATCH * GRP * 4;
  _Float16* q16  = (_Float16*)ws; ws += (size_t)BATCH * HW * CH * 2;
  _Float16* k16  = (_Float16*)ws; ws += (size_t)BATCH * HW * CH * 2;
  _Float16* vt16 = (_Float16*)ws; ws += (size_t)BATCH * HW * CH * 2;
  _Float16* h2   = (_Float16*)ws; ws += (size_t)BATCH * HW * CH * 2;

  cvt_weights_kernel<<<(3 * CH * CH + 255) / 256, 256, 0, stream>>>(
      w_qkv, w_proj, wq16, wp16);
  gn_stats_kernel<<<BATCH * GRP, 256, 0, stream>>>(x, mean, rstd);
  qkv_gemm_kernel<<<dim3(12, 16, BATCH), 256, 0, stream>>>(
      x, gamma, beta, mean, rstd, wq16, b_qkv, q16, k16, vt16);
  attn_kernel<<<dim3(HW / 16, BATCH), 256, 0, stream>>>(q16, k16, vt16, h2);
  proj_kernel<<<dim3(4, 16, BATCH), 256, 0, stream>>>(h2, wp16, b_proj, x, out);
}
